// GCNNodeClassifierNetwork_18975165513738
// MI455X (gfx1250) — matrix operand order repaired and checked by reading, not compiled, not run on hardware
//
#include <hip/hip_runtime.h>
#include <hip/hip_bf16.h>

// Problem constants (from reference)
#define NN   12288   // nodes
#define IND  16      // input dim == num classes
#define HID  256     // hidden
#define NC   16      // classes

typedef __attribute__((ext_vector_type(16))) _Float16 v16h;
typedef __attribute__((ext_vector_type(8)))  _Float16 v8h;
typedef __attribute__((ext_vector_type(8)))  float    v8f;

// ---------------------------------------------------------------- zero deg
__global__ void k_zero(float* __restrict__ p) {
    p[blockIdx.x * 256 + threadIdx.x] = 0.0f;
}

// ------------------------------------------- prologue: A(f32) -> AT(f16) + deg
// 64x64 tiles: coalesced f32 loads, LDS transpose, coalesced f16 stores of A^T,
// and exact column sums (binary values are exact in f16, partials <= 64).
__global__ void __launch_bounds__(256) k_prep(const float* __restrict__ A,
                                              _Float16* __restrict__ AT,
                                              float* __restrict__ deg)
{
    __shared__ _Float16 tile[64][72];   // [col][row], padded
    __shared__ float    csum[64];
    const int t  = threadIdx.x;
    const int r0 = blockIdx.y * 64;
    const int c0 = blockIdx.x * 64;
    if (t < 64) csum[t] = 0.0f;

    const int lr = t >> 4;          // 0..15
    const int lc = (t & 15) * 4;    // 0..60
    #pragma unroll
    for (int rr = 0; rr < 64; rr += 16) {
        const float4 v = *(const float4*)(A + (size_t)(r0 + lr + rr) * NN + c0 + lc);
        tile[lc + 0][lr + rr] = (_Float16)v.x;
        tile[lc + 1][lr + rr] = (_Float16)v.y;
        tile[lc + 2][lr + rr] = (_Float16)v.z;
        tile[lc + 3][lr + rr] = (_Float16)v.w;
    }
    __syncthreads();

    const int oc = t >> 2;          // output row (= column of A), 0..63
    const int ch = (t & 3) * 16;    // 16-half chunk
    float s = 0.0f;
    v8h o0, o1;
    #pragma unroll
    for (int k = 0; k < 8; ++k) { _Float16 h = tile[oc][ch + k];     o0[k] = h; s += (float)h; }
    #pragma unroll
    for (int k = 0; k < 8; ++k) { _Float16 h = tile[oc][ch + 8 + k]; o1[k] = h; s += (float)h; }
    _Float16* dst = AT + (size_t)(c0 + oc) * NN + r0 + ch;
    *(v8h*)dst       = o0;
    *(v8h*)(dst + 8) = o1;
    atomicAdd(&csum[oc], s);        // LDS float atomic (ds_add_f32)
    __syncthreads();
    if (t < 64) atomicAdd(&deg[c0 + t], csum[t]);
}

// ------------------------------------------------------------- d^-1/2 (self loop)
__global__ void k_dis(const float* __restrict__ deg, float* __restrict__ dis) {
    const int i = blockIdx.x * 256 + threadIdx.x;
    dis[i] = rsqrtf(deg[i] + 1.0f);   // +1 self loop; always > 0
}

// ------------------------------ Hs1^T[f][i] = f16( dis[i] * (x @ W1)[i][f] )
__global__ void __launch_bounds__(256) k_h1(const float* __restrict__ x,
                                            const float* __restrict__ W1,
                                            const float* __restrict__ dis,
                                            _Float16* __restrict__ hs1t)
{
    __shared__ float xs[16][17];
    const int ti = threadIdx.x & 15;
    const int tf = threadIdx.x >> 4;
    const int i  = blockIdx.x * 16 + ti;
    const int f  = blockIdx.y * 16 + tf;
    xs[ti][tf] = x[(size_t)i * IND + tf];
    __syncthreads();
    float acc = 0.0f;
    #pragma unroll
    for (int d = 0; d < IND; ++d) acc += xs[ti][d] * W1[d * HID + f];
    hs1t[(size_t)f * NN + i] = (_Float16)(acc * dis[i]);
}

// -------- layer-1 aggregation (WMMA) + fused epilogue (relu, @W2, dis scale)
// Y1^T = Hs1^T x A_hat : A-operand = Hs1^T tile (16f x 32i), B-operand = A^T rows.
// One workgroup: 16 waves x 16 features = all 256 hidden dims of one 16-node tile.
__global__ void __launch_bounds__(512) k_agg1(const _Float16* __restrict__ AT,
                                              const _Float16* __restrict__ hs1t,
                                              const float* __restrict__ dis,
                                              const float* __restrict__ b1,
                                              const float* __restrict__ W2,
                                              _Float16* __restrict__ hs2t)
{
    __shared__ float h1s[HID][16];               // relu(gcn1) tile, 16 KB
    const int j0   = blockIdx.x * 16;
    const int wave = threadIdx.x >> 5;           // f-tile (wave32)
    const int lane = threadIdx.x & 31;
    const int m    = lane & 15;
    const int hf   = lane >> 4;
    const int f0   = wave * 16;
    // A-operand lanes 0-15: K 0-7 & 16-23 ; lanes 16-31: K 8-15 & 24-31
    const _Float16* aRow = hs1t + (size_t)(f0 + m) * NN + 8 * hf;
    // B-operand lane: n = lane&15, K 8*(lane>>4)+0..7 & 16+8*(lane>>4)+0..7 in A^T row j0+n
    const _Float16* bRow = AT + (size_t)(j0 + m) * NN + 8 * hf;

    v8f acc = {0.f, 0.f, 0.f, 0.f, 0.f, 0.f, 0.f, 0.f};
    for (int kb = 0; kb < NN; kb += 32) {
        v8h a0 = *(const v8h*)(aRow + kb);
        v8h a1 = *(const v8h*)(aRow + kb + 16);
        v8h b0 = *(const v8h*)(bRow + kb);
        v8h b1v = *(const v8h*)(bRow + kb + 16);
        v16h aop = __builtin_shufflevector(a0, a1, 0,1,2,3,4,5,6,7,8,9,10,11,12,13,14,15);
        v16h bop = __builtin_shufflevector(b0, b1v, 0,1,2,3,4,5,6,7,8,9,10,11,12,13,14,15);
        __builtin_prefetch(bRow + kb + 512, 0, 0);   // stream A^T (global_prefetch_b8)
        acc = __builtin_amdgcn_wmma_f32_16x16x32_f16(false, aop, false, bop,
                                                     (short)0, acc, false, false);
    }
    // D layout: lane holds node n = lane&15; rows M = v + 8*(lane>>4)
    const float dj = dis[j0 + m];
    #pragma unroll
    for (int v = 0; v < 8; ++v) {
        const int f = f0 + 8 * hf + v;
        const float self = (float)hs1t[(size_t)f * NN + j0 + m];   // A_hat = A + I
        const float y = dj * (acc[v] + self) + b1[f];
        h1s[f][m] = fmaxf(y, 0.0f);                                 // ReLU
    }
    __syncthreads();
    // Hs2^T[c][j] = f16( dis[j] * sum_f h1[j][f] * W2[f][c] )
    if (threadIdx.x < 256) {
        const int c = threadIdx.x >> 4;
        const int j = threadIdx.x & 15;
        float s = 0.0f;
        for (int f = 0; f < HID; ++f) s += h1s[f][j] * W2[f * NC + c];
        hs2t[(size_t)c * NN + j0 + j] = (_Float16)(dis[j0 + j] * s);
    }
}

// -------- layer-2 aggregation (WMMA, split-K over 4 waves) + bias + skip
__global__ void __launch_bounds__(128) k_agg2(const _Float16* __restrict__ AT,
                                              const _Float16* __restrict__ hs2t,
                                              const float* __restrict__ dis,
                                              const float* __restrict__ b2,
                                              const float* __restrict__ xin,
                                              float* __restrict__ obuf)
{
    __shared__ float part[4][8][32];
    const int j0   = blockIdx.x * 16;
    const int wave = threadIdx.x >> 5;
    const int lane = threadIdx.x & 31;
    const int m    = lane & 15;
    const int hf   = lane >> 4;
    const _Float16* aRow = hs2t + (size_t)m * NN + 8 * hf;          // 16 classes
    const _Float16* bRow = AT + (size_t)(j0 + m) * NN + 8 * hf;

    v8f acc = {0.f, 0.f, 0.f, 0.f, 0.f, 0.f, 0.f, 0.f};
    const int kbeg = wave * (NN / 4);
    for (int kk = 0; kk < NN / 4; kk += 32) {
        const int kb = kbeg + kk;
        v8h a0 = *(const v8h*)(aRow + kb);
        v8h a1 = *(const v8h*)(aRow + kb + 16);
        v8h b0 = *(const v8h*)(bRow + kb);
        v8h b1v = *(const v8h*)(bRow + kb + 16);
        v16h aop = __builtin_shufflevector(a0, a1, 0,1,2,3,4,5,6,7,8,9,10,11,12,13,14,15);
        v16h bop = __builtin_shufflevector(b0, b1v, 0,1,2,3,4,5,6,7,8,9,10,11,12,13,14,15);
        __builtin_prefetch(bRow + kb + 512, 0, 0);
        acc = __builtin_amdgcn_wmma_f32_16x16x32_f16(false, aop, false, bop,
                                                     (short)0, acc, false, false);
    }
    #pragma unroll
    for (int v = 0; v < 8; ++v) part[wave][v][lane] = acc[v];
    __syncthreads();
    if (wave == 0) {
        #pragma unroll
        for (int v = 0; v < 8; ++v) {
            float s = part[0][v][lane] + part[1][v][lane]
                    + part[2][v][lane] + part[3][v][lane];
            const int c = 8 * hf + v;
            const int j = j0 + m;
            const float self = (float)hs2t[(size_t)c * NN + j];
            obuf[(size_t)j * NC + c] =
                dis[j] * (s + self) + b2[c] + xin[(size_t)j * NC + c];  // + skip
        }
    }
}

// ----------------------------- softmax over axis 0: per-class max & 1/sumexp
__global__ void __launch_bounds__(256) k_stats(const float* __restrict__ obuf,
                                               float* __restrict__ red)
{
    __shared__ float sm[16][17];
    const int c = threadIdx.x & 15;
    const int r = threadIdx.x >> 4;
    float mx = -3.402823466e38f;
    for (int j = r; j < NN; j += 16) mx = fmaxf(mx, obuf[(size_t)j * NC + c]);
    sm[r][c] = mx;
    __syncthreads();
    float fullmax = sm[0][c];
    #pragma unroll
    for (int k = 1; k < 16; ++k) fullmax = fmaxf(fullmax, sm[k][c]);
    __syncthreads();
    float s = 0.0f;
    for (int j = r; j < NN; j += 16) s += __expf(obuf[(size_t)j * NC + c] - fullmax);
    sm[r][c] = s;
    __syncthreads();
    if (r == 0) {
        float tot = 0.0f;
        #pragma unroll
        for (int k = 0; k < 16; ++k) tot += sm[k][c];
        red[c]      = fullmax;
        red[16 + c] = 1.0f / tot;
    }
}

__global__ void k_write(const float* __restrict__ obuf,
                        const float* __restrict__ red,
                        float* __restrict__ out)
{
    const int idx = blockIdx.x * 256 + threadIdx.x;
    const int c = idx & 15;
    out[idx] = __expf(obuf[idx] - red[c]) * red[16 + c];
}

// --------------------------------------------------------------------------
extern "C" void kernel_launch(void* const* d_in, const int* in_sizes, int n_in,
                              void* d_out, int out_size, void* d_ws, size_t ws_size,
                              hipStream_t stream)
{
    (void)in_sizes; (void)n_in; (void)out_size; (void)ws_size;
    const float* A  = (const float*)d_in[0];
    const float* x  = (const float*)d_in[1];
    const float* W1 = (const float*)d_in[2];
    const float* b1 = (const float*)d_in[3];
    const float* W2 = (const float*)d_in[4];
    const float* b2 = (const float*)d_in[5];
    // d_in[6] sigmoid_param: unused by softmax layer

    char* p = (char*)d_ws;
    _Float16* AT   = (_Float16*)p;  p += (size_t)NN * NN * sizeof(_Float16);  // 302 MB
    float*    deg  = (float*)p;     p += (size_t)NN * sizeof(float);
    float*    dis  = (float*)p;     p += (size_t)NN * sizeof(float);
    _Float16* hs1t = (_Float16*)p;  p += (size_t)HID * NN * sizeof(_Float16); // 6.3 MB
    _Float16* hs2t = (_Float16*)p;  p += (size_t)NC * NN * sizeof(_Float16);
    float*    obuf = (float*)p;     p += (size_t)NN * NC * sizeof(float);
    float*    red  = (float*)p;

    k_zero <<<NN / 256, 256, 0, stream>>>(deg);
    k_prep <<<dim3(NN / 64, NN / 64), 256, 0, stream>>>(A, AT, deg);
    k_dis  <<<NN / 256, 256, 0, stream>>>(deg, dis);
    k_h1   <<<dim3(NN / 16, HID / 16), 256, 0, stream>>>(x, W1, dis, hs1t);
    k_agg1 <<<NN / 16, 512, 0, stream>>>(AT, hs1t, dis, b1, W2, hs2t);
    k_agg2 <<<NN / 16, 128, 0, stream>>>(AT, hs2t, dis, b2, x, obuf);
    k_stats<<<1, 256, 0, stream>>>(obuf, red);
    k_write<<<(NN * NC) / 256, 256, 0, stream>>>(obuf, red, (float*)d_out);
}
